// SCTN_57698590654736
// MI455X (gfx1250) — compile-verified, matrix operand order checked by reading, not run on hardware
//
#include <hip/hip_runtime.h>

typedef __attribute__((ext_vector_type(16))) _Float16 v16h;
typedef __attribute__((ext_vector_type(8)))  _Float16 v8h;
typedef __attribute__((ext_vector_type(8)))  float    v8f;

#define B_   2
#define N_   4096
#define C_   64
#define KNN_ 32

// ---------------------------------------------------------------------------
// WMMA helper: D = A(16x32 f16) x B(32x16 f16) + C(16x16 f32)
// ---------------------------------------------------------------------------
__device__ __forceinline__ v8f wmma_f16f32(v16h a, v16h b, v8f c) {
  return __builtin_amdgcn_wmma_f32_16x16x32_f16(
      /*neg_a=*/false, a, /*neg_b=*/false, b,
      /*c_mod=*/(short)0, c, /*reuse_a=*/false, /*reuse_b=*/false);
}

// Load a 16-row x 32-K f16 operand tile from row-major [row][64] global
// storage into the CDNA5 WMMA A/B lane layout:
//   lane&15 -> row, lane>>4 selects K-chunk interleave
//   halves 0..7 = K koff..koff+7, halves 8..15 = K koff+16..koff+23
__device__ __forceinline__ v16h load_tile(const _Float16* __restrict__ base) {
  int lane = threadIdx.x & 31;
  const _Float16* p = base + (size_t)(lane & 15) * C_ + ((lane >> 4) << 3);
  v8h lo = *(const v8h*)(p);
  v8h hi = *(const v8h*)(p + 16);
  v16h r;
#pragma unroll
  for (int i = 0; i < 8; ++i) { r[i] = lo[i]; r[i + 8] = hi[i]; }
  return r;
}

// Same operand layout, but from an LDS tile with compile-time row stride KP;
// kb selects the 32-wide K-block.
template <int KP>
__device__ __forceinline__ v16h load_tile_lds(const _Float16* base, int kb) {
  int lane = threadIdx.x & 31;
  const _Float16* p = base + (lane & 15) * KP + kb * 32 + ((lane >> 4) << 3);
  v8h lo = *(const v8h*)(p);
  v8h hi = *(const v8h*)(p + 16);
  v16h r;
#pragma unroll
  for (int i = 0; i < 8; ++i) { r[i] = lo[i]; r[i + 8] = hi[i]; }
  return r;
}

// ---------------------------------------------------------------------------
// K0: L2-normalize 64-dim features, convert to f16. One wave per point.
// ---------------------------------------------------------------------------
__global__ void norm_feats_kernel(const float* __restrict__ f, _Float16* __restrict__ o) {
  int lane = threadIdx.x & 31;
  int p = (blockIdx.x << 3) + (threadIdx.x >> 5);
  const float* r = f + (size_t)p * C_;
  float v0 = r[lane], v1 = r[lane + 32];
  float s = v0 * v0 + v1 * v1;
#pragma unroll
  for (int m = 16; m >= 1; m >>= 1) s += __shfl_xor(s, m, 32);
  float inv = rsqrtf(s + 1e-8f);
  _Float16* w = o + (size_t)p * C_;
  w[lane]      = (_Float16)(v0 * inv);
  w[lane + 32] = (_Float16)(v1 * inv);
}

// ---------------------------------------------------------------------------
// K1: column sums of K = exp(-(1-f0.f1)/eps)*[sqdist<100], fused (K never
// materialized). One wave owns a 16-wide column strip and loops all rows.
// ---------------------------------------------------------------------------
__global__ void colsum_kernel(const _Float16* __restrict__ f0h, const _Float16* __restrict__ f1h,
                              const float* __restrict__ pA, const float* __restrict__ pB,
                              const float* __restrict__ eps_in, float* __restrict__ colsum) {
  int lane = threadIdx.x & 31;
  int wv = threadIdx.x >> 5;
  int b = blockIdx.y;
  int m0 = ((blockIdx.x << 3) + wv) << 4;
  int j = lane & 15, h = lane >> 4;
  float eps = __expf(eps_in[0]) + 0.025f;
  float neg_inv_eps = -1.0f / eps;

  const _Float16* fb = f1h + ((size_t)(b * N_) + m0) * C_;
  v16h bt0 = load_tile(fb);
  v16h bt1 = load_tile(fb + 32);
  const float* pj = pB + ((size_t)(b * N_) + m0 + j) * 3;
  float p2x = pj[0], p2y = pj[1], p2z = pj[2];

  float acc = 0.0f;
  for (int n0 = 0; n0 < N_; n0 += 16) {
    const _Float16* fa = f0h + ((size_t)(b * N_) + n0) * C_;
    __builtin_prefetch(fa + 16 * C_, 0, 1);
    v16h at0 = load_tile(fa);
    v16h at1 = load_tile(fa + 32);
    v8f d = {};
    d = wmma_f16f32(at0, bt0, d);
    d = wmma_f16f32(at1, bt1, d);
    const float* pr = pA + ((size_t)(b * N_) + n0 + (h << 3)) * 3;
#pragma unroll
    for (int r = 0; r < 8; ++r) {
      float dx = pr[r * 3 + 0] - p2x, dy = pr[r * 3 + 1] - p2y, dz = pr[r * 3 + 2] - p2z;
      float sq = dx * dx + dy * dy + dz * dz;
      float kv = (sq < 100.0f) ? __expf((1.0f - d[r]) * neg_inv_eps) : 0.0f;
      acc += kv;
    }
  }
  acc += __shfl_xor(acc, 16, 32);  // fold the two row-halves (same column j)
  if (h == 0) colsum[(size_t)b * N_ + m0 + j] = acc;
}

// ---------------------------------------------------------------------------
// K2: Sinkhorn b-vector: b = (prob2/(K^T a + 1e-8))^power, a = 1/N uniform.
// ---------------------------------------------------------------------------
__global__ void bvec_kernel(const float* __restrict__ colsum, const float* g_in,
                            const float* e_in, float* __restrict__ bvec) {
  int i = blockIdx.x * blockDim.x + threadIdx.x;
  if (i >= B_ * N_) return;
  float eps = __expf(e_in[0]) + 0.025f;
  float gam = __expf(g_in[0]);
  float power = gam / (gam + eps);
  float KTa = colsum[i] * (1.0f / N_);
  bvec[i] = powf((1.0f / N_) / (KTa + 1e-8f), power);
}

// ---------------------------------------------------------------------------
// K3: row pass: Kb, weighted pc1 sums, a-update, ot_flow -- all fused.
// One wave owns a 16-row strip and loops all columns.
// ---------------------------------------------------------------------------
__global__ void rowflow_kernel(const _Float16* __restrict__ f0h, const _Float16* __restrict__ f1h,
                               const float* __restrict__ pA, const float* __restrict__ pB,
                               const float* __restrict__ bvec, const float* g_in,
                               const float* e_in, float* __restrict__ flow) {
  int lane = threadIdx.x & 31;
  int wv = threadIdx.x >> 5;
  int b = blockIdx.y;
  int n0 = ((blockIdx.x << 3) + wv) << 4;
  int j = lane & 15, h = lane >> 4;
  float eps = __expf(e_in[0]) + 0.025f;
  float gam = __expf(g_in[0]);
  float power = gam / (gam + eps);
  float neg_inv_eps = -1.0f / eps;

  const _Float16* fa = f0h + ((size_t)(b * N_) + n0) * C_;
  v16h at0 = load_tile(fa);
  v16h at1 = load_tile(fa + 32);

  float px[8], py[8], pz[8];
  const float* pr = pA + ((size_t)(b * N_) + n0 + (h << 3)) * 3;
#pragma unroll
  for (int r = 0; r < 8; ++r) { px[r] = pr[r * 3]; py[r] = pr[r * 3 + 1]; pz[r] = pr[r * 3 + 2]; }

  float kb[8] = {0}, sx[8] = {0}, sy[8] = {0}, sz[8] = {0};
  for (int m0 = 0; m0 < N_; m0 += 16) {
    const _Float16* fb = f1h + ((size_t)(b * N_) + m0) * C_;
    __builtin_prefetch(fb + 16 * C_, 0, 1);
    v16h bt0 = load_tile(fb);
    v16h bt1 = load_tile(fb + 32);
    v8f d = {};
    d = wmma_f16f32(at0, bt0, d);
    d = wmma_f16f32(at1, bt1, d);
    const float* pj = pB + ((size_t)(b * N_) + m0 + j) * 3;
    float qx = pj[0], qy = pj[1], qz = pj[2];
    float bj = bvec[(size_t)b * N_ + m0 + j];
#pragma unroll
    for (int r = 0; r < 8; ++r) {
      float dx = px[r] - qx, dy = py[r] - qy, dz = pz[r] - qz;
      float sq = dx * dx + dy * dy + dz * dz;
      float kv = (sq < 100.0f) ? __expf((1.0f - d[r]) * neg_inv_eps) : 0.0f;
      float kvb = kv * bj;
      kb[r] += kvb; sx[r] += kvb * qx; sy[r] += kvb * qy; sz[r] += kvb * qz;
    }
  }
  // reduce over the 16 columns within each lane-half (butterfly, masks < 16)
#pragma unroll
  for (int r = 0; r < 8; ++r) {
#pragma unroll
    for (int m = 8; m >= 1; m >>= 1) {
      kb[r] += __shfl_xor(kb[r], m, 32); sx[r] += __shfl_xor(sx[r], m, 32);
      sy[r] += __shfl_xor(sy[r], m, 32); sz[r] += __shfl_xor(sz[r], m, 32);
    }
  }
  if (j == 0) {
#pragma unroll
    for (int r = 0; r < 8; ++r) {
      float Kb = kb[r];
      float a = powf((1.0f / N_) / (Kb + 1e-8f), power);
      float rs = a * Kb + 1e-8f;
      int i = n0 + (h << 3) + r;
      float* o = flow + ((size_t)(b * N_) + i) * 3;
      o[0] = a * sx[r] / rs - px[r];
      o[1] = a * sy[r] / rs - py[r];
      o[2] = a * sz[r] / rs - pz[r];
    }
  }
}

// ---------------------------------------------------------------------------
// K4: exact 32-NN (matching jax top_k tie order) - one wave per query point,
// 32 selection rounds with lexicographic (dist,index) wave-min reduction.
// ---------------------------------------------------------------------------
__global__ void knn_kernel(const float* __restrict__ pA, int* __restrict__ knn_out,
                           float* __restrict__ edge) {
  int lane = threadIdx.x & 31;
  int p = (blockIdx.x << 3) + (threadIdx.x >> 5);  // b*N + n
  int b = p >> 12;
  const float* pb = pA + (size_t)(b * N_) * 3;
  float qx = pA[(size_t)p * 3], qy = pA[(size_t)p * 3 + 1], qz = pA[(size_t)p * 3 + 2];
  float prevd = -1.0f; int previ = -1;
  for (int kk = 0; kk < KNN_; ++kk) {
    float bd = 3.4e38f; int bi = 0x7fffffff;
    for (int c = lane; c < N_; c += 32) {
      float dx = pb[c * 3] - qx, dy = pb[c * 3 + 1] - qy, dz = pb[c * 3 + 2] - qz;
      float d = dx * dx + dy * dy + dz * dz;
      bool elig = (d > prevd) || (d == prevd && c > previ);
      if (elig && (d < bd || (d == bd && c < bi))) { bd = d; bi = c; }
    }
#pragma unroll
    for (int m = 16; m >= 1; m >>= 1) {
      float od = __shfl_xor(bd, m, 32);
      int oi = __shfl_xor(bi, m, 32);
      if (od < bd || (od == bd && oi < bi)) { bd = od; bi = oi; }
    }
    prevd = bd; previ = bi;
    if (lane == 0) {
      knn_out[(size_t)p * KNN_ + kk] = bi;
      float* e = edge + ((size_t)p * KNN_ + kk) * 3;
      e[0] = pb[bi * 3] - qx; e[1] = pb[bi * 3 + 1] - qy; e[2] = pb[bi * 3 + 2] - qz;
    }
  }
}

// ---------------------------------------------------------------------------
// K5: set conv. Stage 1 (gather -> MLP -> max over 32 neighbors) runs on
// WMMA: per point, the 32 gathered rows form two 16xKP f16 A-tiles staged in
// LDS in operand layout; w1^T is staged once per block as the B operand.
// Bias + leaky-ReLU + row-max are applied directly on the D accumulator
// layout. Dense stages 2/3 stay lane-parallel f32 VALU.
// ---------------------------------------------------------------------------
template <int CIN, int H1, int H2, int H3>
__global__ __launch_bounds__(256) void setconv_kernel(
    const float* __restrict__ feat, const int* __restrict__ knn_in,
    const float* __restrict__ edge,
    const float* __restrict__ w1, const float* __restrict__ b1,
    const float* __restrict__ w2, const float* __restrict__ b2,
    const float* __restrict__ w3, const float* __restrict__ b3,
    float* __restrict__ out) {
  constexpr int KH = CIN + 3;              // gathered h-vector length
  constexpr int KP = (KH + 31) & ~31;      // padded to f16-WMMA K multiple
  constexpr int NK = KP / 32;              // K-blocks per tile
  constexpr int NCT = H1 / 16;             // 16-wide output-channel tiles
  constexpr int T2 = H2 / 32, T3 = H3 / 32;

  __shared__ _Float16 wT[H1 * KP];         // w1 transposed [col][k], f16
  __shared__ _Float16 hb[8 * 16 * KP];     // per-wave 16-row A tile
  __shared__ float s1[8][H1];
  __shared__ float s2[8][H2];

  int tid = threadIdx.x;
  int lane = tid & 31, wv = tid >> 5;
  int p = (blockIdx.x << 3) + wv;          // b*N + n
  int bbase = p & ~(N_ - 1);
  const int* kn = knn_in + (size_t)p * KNN_;
  const float* eg = edge + (size_t)p * KNN_ * 3;

  // Stage w1 -> LDS (transposed, f16, zero-padded K)
  for (int i = tid; i < H1 * KP; i += 256) {
    int col = i / KP, c = i - col * KP;
    wT[i] = (_Float16)((c < KH) ? w1[c * H1 + col] : 0.0f);
  }
  __syncthreads();

  float cmax[NCT];
#pragma unroll
  for (int ct = 0; ct < NCT; ++ct) cmax[ct] = -3.4e38f;

  _Float16* myh = &hb[wv * 16 * KP];
  int row = lane & 15;
  int k0 = (lane >> 4) * (KP / 2);

  for (int half = 0; half < 2; ++half) {
    // gather my neighbor row (features ++ edge, zero-padded) into LDS as f16
    int kk = (half << 4) + row;
    int nb = bbase + kn[kk];
    const float* fr = feat + (size_t)nb * CIN;
    const float* er = eg + kk * 3;
#pragma unroll
    for (int kc = 0; kc < KP / 16; ++kc) {   // (KP/2) values, 8 per b128 store
      v8h chunk;
#pragma unroll
      for (int u = 0; u < 8; ++u) {
        int k = k0 + kc * 8 + u;
        float v = (k < CIN) ? fr[k] : ((k < KH) ? er[k - CIN] : 0.0f);
        chunk[u] = (_Float16)v;
      }
      *(v8h*)&myh[row * KP + k0 + kc * 8] = chunk;
    }
    __syncthreads();  // uniform: all waves run identical trip counts

    v16h at[NK];
#pragma unroll
    for (int kb = 0; kb < NK; ++kb) at[kb] = load_tile_lds<KP>(myh, kb);

#pragma unroll
    for (int ct = 0; ct < NCT; ++ct) {
      v8f d = {};
#pragma unroll
      for (int kb = 0; kb < NK; ++kb) {
        v16h bt = load_tile_lds<KP>(&wT[ct * 16 * KP], kb);
        d = wmma_f16f32(at[kb], bt, d);
      }
      float bias = b1[ct * 16 + (lane & 15)];
      float cm = -3.4e38f;
#pragma unroll
      for (int r = 0; r < 8; ++r) {
        float a = d[r] + bias;
        a = fmaxf(a, 0.1f * a);              // leaky relu (slope .1)
        cm = fmaxf(cm, a);
      }
      cm = fmaxf(cm, __shfl_xor(cm, 16, 32));  // fold row-halves of the tile
      cmax[ct] = fmaxf(cmax[ct], cm);
    }
    __syncthreads();  // hb reused by next half
  }
  if (lane < 16) {
#pragma unroll
    for (int ct = 0; ct < NCT; ++ct) s1[wv][ct * 16 + lane] = cmax[ct];
  }
  __syncthreads();

  // ---- stage 2: dense H1 -> H2, leaky relu ----
  float a2[T2];
#pragma unroll
  for (int t = 0; t < T2; ++t) a2[t] = b2[t * 32 + lane];
  for (int c = 0; c < H1; ++c) {
    float hv = s1[wv][c];
#pragma unroll
    for (int t = 0; t < T2; ++t) a2[t] = fmaf(hv, w2[c * H2 + t * 32 + lane], a2[t]);
  }
#pragma unroll
  for (int t = 0; t < T2; ++t) s2[wv][t * 32 + lane] = fmaxf(a2[t], 0.1f * a2[t]);
  __syncthreads();

  // ---- stage 3: dense H2 -> H3, leaky relu ----
  float a3[T3];
#pragma unroll
  for (int t = 0; t < T3; ++t) a3[t] = b3[t * 32 + lane];
  for (int c = 0; c < H2; ++c) {
    float hv = s2[wv][c];
#pragma unroll
    for (int t = 0; t < T3; ++t) a3[t] = fmaf(hv, w3[c * H3 + t * 32 + lane], a3[t]);
  }
#pragma unroll
  for (int t = 0; t < T3; ++t)
    out[(size_t)p * H3 + t * 32 + lane] = fmaxf(a3[t], 0.1f * a3[t]);
}

// ---------------------------------------------------------------------------
// K6: final FC 128->3 + residual with ot_flow.
// ---------------------------------------------------------------------------
__global__ void final_kernel(const float* __restrict__ x, const float* __restrict__ flow,
                             const float* __restrict__ wfc, const float* __restrict__ bfc,
                             float* __restrict__ out) {
  int tid = blockIdx.x * blockDim.x + threadIdx.x;
  if (tid >= B_ * N_ * 3) return;
  int p = tid / 3, c = tid - p * 3;
  float s = bfc[c];
  const float* xr = x + (size_t)p * 128;
#pragma unroll 8
  for (int k = 0; k < 128; ++k) s = fmaf(xr[k], wfc[k * 3 + c], s);
  out[tid] = flow[tid] + s;
}

// ---------------------------------------------------------------------------
extern "C" void kernel_launch(void* const* d_in, const int* in_sizes, int n_in,
                              void* d_out, int out_size, void* d_ws, size_t ws_size,
                              hipStream_t stream) {
  const float* pc0 = (const float*)d_in[0];
  const float* pc1 = (const float*)d_in[1];
  const float* f0  = (const float*)d_in[2];
  const float* f1  = (const float*)d_in[3];
  const float* gamma   = (const float*)d_in[4];
  const float* epsilon = (const float*)d_in[5];
  const float *w1_1 = (const float*)d_in[6],  *b1_1 = (const float*)d_in[7];
  const float *w1_2 = (const float*)d_in[8],  *b1_2 = (const float*)d_in[9];
  const float *w1_3 = (const float*)d_in[10], *b1_3 = (const float*)d_in[11];
  const float *w2_1 = (const float*)d_in[12], *b2_1 = (const float*)d_in[13];
  const float *w2_2 = (const float*)d_in[14], *b2_2 = (const float*)d_in[15];
  const float *w2_3 = (const float*)d_in[16], *b2_3 = (const float*)d_in[17];
  const float *w3_1 = (const float*)d_in[18], *b3_1 = (const float*)d_in[19];
  const float *w3_2 = (const float*)d_in[20], *b3_2 = (const float*)d_in[21];
  const float *w3_3 = (const float*)d_in[22], *b3_3 = (const float*)d_in[23];
  const float *wfc  = (const float*)d_in[24], *bfc  = (const float*)d_in[25];
  float* out = (float*)d_out;

  const int BN = B_ * N_;
  char* wsp = (char*)d_ws;
  auto alloc = [&](size_t bytes) {
    char* r = wsp;
    wsp += (bytes + 255) & ~(size_t)255;
    return (void*)r;
  };
  _Float16* f0h  = (_Float16*)alloc((size_t)BN * C_ * 2);
  _Float16* f1h  = (_Float16*)alloc((size_t)BN * C_ * 2);
  float* colsum  = (float*)alloc((size_t)BN * 4);
  float* bvec    = (float*)alloc((size_t)BN * 4);
  float* flow    = (float*)alloc((size_t)BN * 3 * 4);
  int*   knn     = (int*)  alloc((size_t)BN * KNN_ * 4);
  float* edge    = (float*)alloc((size_t)BN * KNN_ * 3 * 4);
  float* x1      = (float*)alloc((size_t)BN * 128 * 4);
  float* x2      = (float*)alloc((size_t)BN * 128 * 4);

  norm_feats_kernel<<<BN / 8, 256, 0, stream>>>(f0, f0h);
  norm_feats_kernel<<<BN / 8, 256, 0, stream>>>(f1, f1h);
  colsum_kernel<<<dim3(N_ / 128, B_), 256, 0, stream>>>(f0h, f1h, pc0, pc1, epsilon, colsum);
  bvec_kernel<<<(BN + 255) / 256, 256, 0, stream>>>(colsum, gamma, epsilon, bvec);
  rowflow_kernel<<<dim3(N_ / 128, B_), 256, 0, stream>>>(f0h, f1h, pc0, pc1, bvec, gamma,
                                                         epsilon, flow);
  knn_kernel<<<BN / 8, 256, 0, stream>>>(pc0, knn, edge);
  setconv_kernel<3, 32, 32, 32><<<BN / 8, 256, 0, stream>>>(
      flow, knn, edge, w1_1, b1_1, w1_2, b1_2, w1_3, b1_3, x1);
  setconv_kernel<32, 64, 64, 64><<<BN / 8, 256, 0, stream>>>(
      x1, knn, edge, w2_1, b2_1, w2_2, b2_2, w2_3, b2_3, x2);
  setconv_kernel<64, 128, 128, 128><<<BN / 8, 256, 0, stream>>>(
      x2, knn, edge, w3_1, b3_1, w3_2, b3_2, w3_3, b3_3, x1);
  final_kernel<<<(BN * 3 + 255) / 256, 256, 0, stream>>>(x1, flow, wfc, bfc, out);

  (void)in_sizes; (void)n_in; (void)out_size; (void)ws_size;
}
